// GCN_31413390803463
// MI455X (gfx1250) — matrix operand order repaired and checked by reading, not compiled, not run on hardware
//
#include <hip/hip_runtime.h>

#define NFEAT 256
#define NHID 128
#define NCLASS 40
#define BN_EPS 1e-5f

typedef __attribute__((ext_vector_type(16))) __bf16 v16bf;
typedef __attribute__((ext_vector_type(8)))  __bf16 v8bf;
typedef __attribute__((ext_vector_type(8)))  float  v8f;

// ---------------------------------------------------------------- utilities
__global__ void fill_f32(float* __restrict__ p, long long n, float v) {
  long long i = (long long)blockIdx.x * blockDim.x + threadIdx.x;
  if (i < n) p[i] = v;
}

// deg[dst] += ew  (deg pre-filled with 1.0 for the self loop)
__global__ void deg_accum(const int* __restrict__ dst, const float* __restrict__ ew,
                          float* __restrict__ deg, int e) {
  int i = blockIdx.x * blockDim.x + threadIdx.x;
  if (i < e) atomicAdd(&deg[dst[i]], ew[i]);
}

// deg -> d^{-1/2} in place
__global__ void dinv_k(float* __restrict__ deg, int n) {
  int i = blockIdx.x * blockDim.x + threadIdx.x;
  if (i < n) { float d = deg[i]; deg[i] = d > 0.f ? rsqrtf(d) : 0.f; }
}

__global__ void norm_k(const int* __restrict__ src, const int* __restrict__ dst,
                       const float* __restrict__ ew, const float* __restrict__ dinv,
                       float* __restrict__ norm, int e) {
  int i = blockIdx.x * blockDim.x + threadIdx.x;
  if (i < e) norm[i] = dinv[src[i]] * ew[i] * dinv[dst[i]];
}

// ---------------------------------------------------- weight fragment packer
// Packs W[K,Ncols] (row-major f32) into the exact per-lane WMMA B-fragment
// image (bf16, zero-padded cols), ordered [chunk][tnTile][lane][16 elems] so
// the GEMM can flat-copy it into LDS and read one v16bf per lane.
__global__ void convert_w_frag(const float* __restrict__ W, int Ncols, int ntiles,
                               __bf16* __restrict__ out, int total) {
  int idx = blockIdx.x * blockDim.x + threadIdx.x;   // over chunks*ntiles*32*16
  if (idx >= total) return;
  int j    = idx & 15;
  int lane = (idx >> 4) & 31;
  int t    = (idx >> 9) % ntiles;
  int c    = (idx >> 9) / ntiles;
  int col  = t * 16 + (lane & 15);
  int k    = c * 32 + (lane >> 4) * 8 + (j & 7) + (j >> 3) * 16;
  float w  = (col < Ncols) ? W[(size_t)k * Ncols + col] : 0.0f;
  out[idx] = (__bf16)w;
}

// ------------------------------------------------------------- WMMA GEMM
// C[M,NCOLS] = A[M,K](lda) @ W (+bias). One wave per 16x16 tile; the 8 waves
// of a block share one 16-row A strip when NCOLS==128. W fragments staged
// into LDS via global_load_async_to_lds_b128. ABF16 selects bf16 A loads
// (2x b128, no cvt) vs f32 A loads with on-the-fly conversion.
template <int KDIM, int NCOLS, bool ABF16>
__global__ void __launch_bounds__(256)
gcn_wmma_gemm(const void* __restrict__ Av, int lda,
              const __bf16* __restrict__ wfrag,
              const float* __restrict__ bias,
              float* __restrict__ C, int ldc, int total_tiles) {
  constexpr int NT     = (NCOLS + 15) / 16;
  constexpr int CHUNKS = KDIM / 32;
  constexpr int B16    = CHUNKS * NT * 64;   // #16-byte units of the W image
  extern __shared__ char smem[];

  // ---- async flat copy: global fragment image -> LDS (16B per lane-op) ----
  for (int idx = threadIdx.x; idx < B16; idx += 256) {
    int off = idx * 16;
    asm volatile("global_load_async_to_lds_b128 %0, %1, %2"
                 :: "v"(off), "v"(off), "s"(wfrag) : "memory");
  }
  asm volatile("s_wait_asynccnt 0x0" ::: "memory");
  __syncthreads();

  int wave = threadIdx.x >> 5;
  int tile = blockIdx.x * 8 + wave;
  if (tile >= total_tiles) return;
  int tm   = (tile / NT) << 4;
  int tnT  = tile % NT;
  int lane = threadIdx.x & 31;
  int half = lane >> 4;          // 0: lanes 0-15, 1: lanes 16-31
  int l    = lane & 15;

  const v16bf* Wl = (const v16bf*)smem;
  const float*  Af = (const float*)Av  + (size_t)(tm + l) * lda;
  const __bf16* Ab = (const __bf16*)Av + (size_t)(tm + l) * lda;

  v8f acc = {};
#pragma unroll
  for (int c = 0; c < CHUNKS; ++c) {
    int abase = c * 32 + half * 8;
    v16bf a;
    if constexpr (ABF16) {
      v8bf lo = *(const v8bf*)(Ab + abase);
      v8bf hi = *(const v8bf*)(Ab + abase + 16);
      a = __builtin_shufflevector(lo, hi, 0, 1, 2, 3, 4, 5, 6, 7,
                                  8, 9, 10, 11, 12, 13, 14, 15);
      __builtin_prefetch(Ab + abase + 32, 0, 1);    // global_prefetch_b8
    } else {
      __builtin_prefetch(Af + abase + 32, 0, 1);
#pragma unroll
      for (int j = 0; j < 8; ++j) {
        a[j]     = (__bf16)Af[abase + j];
        a[j + 8] = (__bf16)Af[abase + 16 + j];
      }
    }
    v16bf b = Wl[(c * NT + tnT) * 32 + lane];       // 2x ds_load_b128
    acc = __builtin_amdgcn_wmma_f32_16x16x32_bf16(
        /*neg_a=*/false, a, /*neg_b=*/false, b,
        /*c_mod=*/(short)0, acc, /*reuse_a=*/false, /*reuse_b=*/false);
  }

  int col = tnT * 16 + l;
  if ((NCOLS % 16 == 0) || col < NCOLS) {
    float bv = bias ? bias[col] : 0.0f;
#pragma unroll
    for (int r = 0; r < 8; ++r)
      C[(size_t)(tm + half * 8 + r) * ldc + col] = acc[r] + bv;
  }
}

// ------------------------------------------------------------- aggregation
// agg[i, c4] = b[c4] + dinv[i]^2 * lin[i, c4]  (float4-vectorized: 32 thr/row)
__global__ void agg_init4(const float4* __restrict__ lin4, const float* __restrict__ dinv,
                          const float4* __restrict__ bias4, float4* __restrict__ agg4,
                          long long n32) {
  long long gid = (long long)blockIdx.x * blockDim.x + threadIdx.x;
  if (gid >= n32) return;
  int c4 = (int)(gid & 31);
  long long i = gid >> 5;
  float dv = dinv[i];
  float s = dv * dv;
  float4 v = lin4[gid];
  float4 b = bias4[c4];
  float4 o;
  o.x = b.x + s * v.x; o.y = b.y + s * v.y;
  o.z = b.z + s * v.z; o.w = b.w + s * v.w;
  agg4[gid] = o;
}

// agg[dst] += norm[e]*lin[src] : 32 threads/edge, b128 gather + 4 f32 atomics
__global__ void edge_scatter4(const float4* __restrict__ lin4, const int* __restrict__ src,
                              const int* __restrict__ dst, const float* __restrict__ norm,
                              float* __restrict__ agg, long long e32) {
  long long gid = (long long)blockIdx.x * blockDim.x + threadIdx.x;
  if (gid >= e32) return;
  long long ed = gid >> 5;
  int c4 = (int)(gid & 31);
  float nm = norm[ed];
  int s = src[ed], d = dst[ed];
  float4 v = lin4[(size_t)s * 32 + c4];
  float* ap = agg + (size_t)d * NHID + c4 * 4;
  atomicAdd(ap + 0, nm * v.x);
  atomicAdd(ap + 1, nm * v.y);
  atomicAdd(ap + 2, nm * v.z);
  atomicAdd(ap + 3, nm * v.w);
}

// ------------------------------------------------------------- batch norm
__global__ void bn_reduce(const float* __restrict__ h, int n, int rows, int relu,
                          float* __restrict__ sums, float* __restrict__ sumsq) {
  int c  = threadIdx.x;              // 128 threads, one per column
  int r0 = blockIdx.x * rows;
  int r1 = r0 + rows; if (r1 > n) r1 = n;
  float s = 0.f, q = 0.f;
  for (int r = r0; r < r1; ++r) {
    float v = h[(size_t)r * NHID + c];
    if (relu) v = fmaxf(v, 0.f);
    s += v; q += v * v;
  }
  atomicAdd(&sums[c], s);
  atomicAdd(&sumsq[c], q);
}

__global__ void bn_finalize(const float* __restrict__ sums, const float* __restrict__ sumsq,
                            const float* __restrict__ g, const float* __restrict__ beta,
                            int n, float* __restrict__ scale, float* __restrict__ shift) {
  int c = threadIdx.x;
  float invn = 1.0f / (float)n;
  float m   = sums[c] * invn;
  float var = sumsq[c] * invn - m * m;      // biased batch variance
  float sc  = g[c] * rsqrtf(var + BN_EPS);
  scale[c] = sc;
  shift[c] = beta[c] - m * sc;
}

// writes f32 embed slice (output) AND bf16 activation copy for the next GEMM
__global__ void bn_apply(const float* __restrict__ h, long long n128, int relu,
                         const float* __restrict__ scale, const float* __restrict__ shift,
                         float* __restrict__ outf, __bf16* __restrict__ outb, int ldo) {
  long long gid = (long long)blockIdx.x * blockDim.x + threadIdx.x;
  if (gid >= n128) return;
  long long i = gid >> 7;
  int c = (int)(gid & (NHID - 1));
  float v = h[gid];
  if (relu) v = fmaxf(v, 0.f);
  float r = v * scale[c] + shift[c];
  outf[i * (long long)ldo + c] = r;
  outb[i * (long long)ldo + c] = (__bf16)r;
}

// ------------------------------------------------------------- softmax
__global__ void softmax_k(const float* __restrict__ logits, float* __restrict__ out, int n) {
  int i = blockIdx.x * blockDim.x + threadIdx.x;
  if (i >= n) return;
  const float* row = logits + (size_t)i * NCLASS;
  float mx = row[0];
#pragma unroll
  for (int j = 1; j < NCLASS; ++j) mx = fmaxf(mx, row[j]);
  float ex[NCLASS]; float s = 0.f;
#pragma unroll
  for (int j = 0; j < NCLASS; ++j) { ex[j] = __expf(row[j] - mx); s += ex[j]; }
  float inv = 1.0f / s;
  float* o = out + (size_t)i * NCLASS;
#pragma unroll
  for (int j = 0; j < NCLASS; ++j) o[j] = ex[j] * inv;
}

// ------------------------------------------------------------- launcher
extern "C" void kernel_launch(void* const* d_in, const int* in_sizes, int n_in,
                              void* d_out, int out_size, void* d_ws, size_t ws_size,
                              hipStream_t stream) {
  (void)n_in; (void)out_size; (void)ws_size;
  const float* x    = (const float*)d_in[0];
  const int*   ei   = (const int*)d_in[1];
  const float* ew   = (const float*)d_in[2];
  const float* Ws[3]  = {(const float*)d_in[3],  (const float*)d_in[7],  (const float*)d_in[11]};
  const float* bs[3]  = {(const float*)d_in[4],  (const float*)d_in[8],  (const float*)d_in[12]};
  const float* gs[3]  = {(const float*)d_in[5],  (const float*)d_in[9],  (const float*)d_in[13]};
  const float* bts[3] = {(const float*)d_in[6],  (const float*)d_in[10], (const float*)d_in[14]};
  const float* linW = (const float*)d_in[15];
  const float* linb = (const float*)d_in[16];

  const int n = in_sizes[0] / NFEAT;     // 100000
  const int e = in_sizes[1] / 2;         // 3200000
  const int* srcI = ei;
  const int* dstI = ei + e;

  // ws: wfrag(64KB) | dinv[n] | norm[e] | lin[n*128] | agg[n*128] | stats | hbf
  __bf16* wfrag = (__bf16*)d_ws;
  float* fbase = (float*)((char*)d_ws + 65536);
  float* dinv  = fbase;
  float* norm  = dinv + n;
  float* lin   = norm + e;
  float* agg   = lin + (size_t)n * NHID;
  float* sums  = agg + (size_t)n * NHID;
  float* sumsq = sums + NHID;
  float* scale = sumsq + NHID;
  float* shift = scale + NHID;
  __bf16* hbf  = (__bf16*)(shift + NHID);     // [n, 384] bf16 activations

  // output layout: logits[n*40] | softmax[n*40] | embed[n*384]
  float* logits = (float*)d_out;
  float* soft   = logits + (size_t)n * NCLASS;
  float* embed  = soft + (size_t)n * NCLASS;
  const int lde = 3 * NHID;

  auto cdiv = [](long long a, long long b) { return (int)((a + b - 1) / b); };

  // --- gcn_norm ---
  fill_f32<<<cdiv(n, 256), 256, 0, stream>>>(dinv, n, 1.0f);   // self-loop weight
  deg_accum<<<cdiv(e, 256), 256, 0, stream>>>(dstI, ew, dinv, e);
  dinv_k<<<cdiv(n, 256), 256, 0, stream>>>(dinv, n);
  norm_k<<<cdiv(e, 256), 256, 0, stream>>>(srcI, dstI, ew, dinv, norm, e);

  const long long n128 = (long long)n * NHID;
  const long long n32  = (long long)n * 32;
  const long long e32  = (long long)e * 32;
  const int mtiles = n / 16;             // 100000 % 16 == 0

  for (int layer = 0; layer < 3; ++layer) {
    int relu = (layer < 2) ? 1 : 0;
    const int total = mtiles * (NHID / 16);
    if (layer == 0) {
      const int welems = (NFEAT / 32) * 8 * 512;               // 32768
      convert_w_frag<<<cdiv(welems, 256), 256, 0, stream>>>(Ws[0], NHID, 8, wfrag, welems);
      gcn_wmma_gemm<NFEAT, NHID, false><<<cdiv(total, 8), 256, (size_t)welems * 2, stream>>>(
          x, NFEAT, wfrag, nullptr, lin, NHID, total);
    } else {
      const int welems = (NHID / 32) * 8 * 512;                // 16384
      convert_w_frag<<<cdiv(welems, 256), 256, 0, stream>>>(Ws[layer], NHID, 8, wfrag, welems);
      gcn_wmma_gemm<NHID, NHID, true><<<cdiv(total, 8), 256, (size_t)welems * 2, stream>>>(
          hbf + (layer - 1) * NHID, lde, wfrag, nullptr, lin, NHID, total);
    }
    agg_init4<<<cdiv(n32, 256), 256, 0, stream>>>(
        (const float4*)lin, dinv, (const float4*)bs[layer], (float4*)agg, n32);
    edge_scatter4<<<cdiv(e32, 256), 256, 0, stream>>>(
        (const float4*)lin, srcI, dstI, norm, agg, e32);
    fill_f32<<<1, 256, 0, stream>>>(sums, 2 * NHID, 0.0f);     // sums+sumsq contiguous
    bn_reduce<<<cdiv(n, 512), NHID, 0, stream>>>(agg, n, 512, relu, sums, sumsq);
    bn_finalize<<<1, NHID, 0, stream>>>(sums, sumsq, gs[layer], bts[layer], n, scale, shift);
    bn_apply<<<cdiv(n128, 256), 256, 0, stream>>>(
        agg, n128, relu, scale, shift, embed + layer * NHID, hbf + layer * NHID, lde);
  }

  // final linear: K=384 bf16 activations, 40 cols -> 3 n-tiles, bias fused
  {
    const int welems = (384 / 32) * 3 * 512;                   // 18432
    convert_w_frag<<<cdiv(welems, 256), 256, 0, stream>>>(linW, NCLASS, 3, wfrag, welems);
    const int total = mtiles * 3;
    gcn_wmma_gemm<384, NCLASS, true><<<cdiv(total, 8), 256, (size_t)welems * 2, stream>>>(
        hbf, lde, wfrag, linb, logits, NCLASS, total);
  }
  softmax_k<<<cdiv(n, 256), 256, 0, stream>>>(logits, soft, n);
}
